// AttentionEnsemble_83829171683524
// MI455X (gfx1250) — matrix operand order repaired and checked by reading, not compiled, not run on hardware
//
#include <hip/hip_runtime.h>
#include <hip/hip_bf16.h>

#define DEVINL __device__ __forceinline__

typedef unsigned short u16t;
typedef __attribute__((ext_vector_type(16))) __bf16 v16bf;
typedef __attribute__((ext_vector_type(8)))  float  v8f;
typedef __attribute__((ext_vector_type(8)))  unsigned short us8;
typedef __attribute__((ext_vector_type(4)))  int    v4i;

union BFrag { v16bf v; u16t u[16]; };

constexpr int B_ = 2, S_ = 2048, D_ = 1024, H_ = 16, HD_ = 64;

// ---- CDNA5 async global->LDS path (guarded; falls back to sync copy) ----
#if defined(__HIP_DEVICE_COMPILE__) && defined(__gfx1250__) && \
    __has_builtin(__builtin_amdgcn_global_load_async_to_lds_b128)
#define USE_ASYNC_LDS 1
#else
#define USE_ASYNC_LDS 0
#endif

#if USE_ASYNC_LDS && __has_builtin(__builtin_amdgcn_s_wait_asynccnt)
#define ASYNC_WAIT(n) __builtin_amdgcn_s_wait_asynccnt(n)
#elif USE_ASYNC_LDS
#define ASYNC_WAIT(n) asm volatile("s_wait_asynccnt %0" :: "i"(n) : "memory")
#else
#define ASYNC_WAIT(n) ((void)0)
#endif

#if USE_ASYNC_LDS
typedef __attribute__((address_space(1))) v4i gv4i;   // global 128-bit
typedef __attribute__((address_space(3))) v4i lv4i;   // LDS 128-bit
#endif

// Copy 16 bytes global -> LDS (async when available).
DEVINL void cp16(u16t* lds, const u16t* g) {
#if USE_ASYNC_LDS
  __builtin_amdgcn_global_load_async_to_lds_b128(
      (gv4i*)(u16t*)g, (lv4i*)lds, 0, 0);
#else
  *(us8*)lds = *(const us8*)g;
#endif
}

DEVINL u16t f32_bf16(float f) {
  union { float f; unsigned u; } c; c.f = f;
  unsigned r = c.u + 0x7FFFu + ((c.u >> 16) & 1u);   // round-to-nearest-even
  return (u16t)(r >> 16);
}

DEVINL v8f vzero() { v8f z = {0.f,0.f,0.f,0.f,0.f,0.f,0.f,0.f}; return z; }

DEVINL v8f wmma_bf16(const BFrag& a, const BFrag& b, v8f c) {
  return __builtin_amdgcn_wmma_f32_16x16x32_bf16(false, a.v, false, b.v,
                                                 (short)0, c, false, false);
}

// A-fragment (16x32 slice of row-major source, stride lda elems).
// ISA: lanes 0-15 row=lane, u[0..7]=K0..7, u[8..15]=K16..23;
//      lanes 16-31 row=lane-16, u[0..7]=K8..15, u[8..15]=K24..31.
DEVINL void load_a(BFrag& f, const u16t* a, int lda, int lane) {
  int row = lane & 15;
  int k0  = (lane < 16) ? 0 : 8;
  const u16t* p = a + row * lda + k0;
#pragma unroll
  for (int i = 0; i < 8; i++) { f.u[i] = p[i]; f.u[8 + i] = p[16 + i]; }
}

// B-fragment from K-contiguous storage [n][k] (stride ldk).
// ISA: N = lane&15; lanes 0-15: u[0..7]=K0..7, u[8..15]=K16..23; lanes 16-31: u[0..7]=K8..15, u[8..15]=K24..31.
DEVINL void load_b_kc(BFrag& f, const u16t* b, int ldk, int lane) {
  int n  = lane & 15;
  int k0 = (lane < 16) ? 0 : 8;
  const u16t* p = b + n * ldk + k0;
#pragma unroll
  for (int i = 0; i < 8; i++) { f.u[i] = p[i]; f.u[8 + i] = p[16 + i]; }
}

DEVINL float redmax16(float v) {
#pragma unroll
  for (int m = 1; m < 16; m <<= 1) v = fmaxf(v, __shfl_xor(v, m, 32));
  return v;
}
DEVINL float redsum16(float v) {
#pragma unroll
  for (int m = 1; m < 16; m <<= 1) v += __shfl_xor(v, m, 32);
  return v;
}

// ---------------- Stage 0a: fp32 -> bf16 (same layout) ----------------
__global__ void k_cvt(const float* __restrict__ in, u16t* __restrict__ out, int n) {
  int i = blockIdx.x * blockDim.x + threadIdx.x;
  if (i < n) out[i] = f32_bf16(in[i]);
}

// ---------------- Stage 0b: fp32 -> bf16 with per-matrix transpose --------
// in: nmat matrices RxC (row-major) -> out: nmat matrices CxR (row-major).
__global__ void k_cvt_t(const float* __restrict__ in, u16t* __restrict__ out,
                        int R, int C, int n) {
  int i = blockIdx.x * blockDim.x + threadIdx.x;
  if (i >= n) return;
  int rc = R * C;
  int m = i / rc, rem = i - m * rc;
  int r = rem / C, c = rem - r * C;
  out[(size_t)m * rc + (size_t)c * R + r] = f32_bf16(in[i]);
}

// ---------------- Stage 1: QKV projection ----------------
// grid (B*H, S/128), 256 threads (8 waves); wave w -> 16 rows.
// Weights pre-transposed wT[h][e][d]; V written transposed [bh][e][s].
// Double-buffered async global->LDS pipeline: tile i+1 in flight during tile i.
__global__ __launch_bounds__(256) void k_qkv(
    const u16t* __restrict__ xb,
    const u16t* __restrict__ wqT, const u16t* __restrict__ wkT, const u16t* __restrict__ wvT,
    u16t* __restrict__ qb, u16t* __restrict__ kb, u16t* __restrict__ vbT) {
  __shared__ u16t lx[2][128 * 32];      // 16 KB
  __shared__ u16t lw[2][3][64 * 32];    // 24 KB, [e][d] K-contiguous tiles
  int t = threadIdx.x, lane = t & 31, w = t >> 5;
  int bh = blockIdx.x, h = bh & (H_ - 1), b = bh >> 4;
  int s0 = blockIdx.y * 128;
  const u16t* wmsT[3] = { wqT + (size_t)h * HD_ * D_,
                          wkT + (size_t)h * HD_ * D_,
                          wvT + (size_t)h * HD_ * D_ };
  v8f acc[3][4];
#pragma unroll
  for (int m = 0; m < 3; m++)
#pragma unroll
    for (int n = 0; n < 4; n++) acc[m][n] = vzero();

  int xr = t >> 1, xc = (t & 1) * 16;         // x-tile staging coords
  int f0 = t * 8, we = f0 >> 5, wd = f0 & 31; // W-tile staging coords

  auto issue = [&](int d0, int buf) {          // 5 async ops per thread
    const u16t* xsrc = xb + (size_t)(b * S_ + s0 + xr) * D_ + d0 + xc;
    cp16(&lx[buf][xr * 32 + xc], xsrc);
    cp16(&lx[buf][xr * 32 + xc + 8], xsrc + 8);
#pragma unroll
    for (int m = 0; m < 3; m++)
      cp16(&lw[buf][m][we * 32 + wd], wmsT[m] + (size_t)we * D_ + d0 + wd);
  };

  constexpr int NIT = D_ / 32;
  issue(0, 0);
  for (int it = 0; it < NIT; it++) {
    int buf = it & 1;
    if (it + 1 < NIT) { issue((it + 1) * 32, (it + 1) & 1); ASYNC_WAIT(5); }
    else              { ASYNC_WAIT(0); }
    __syncthreads();                       // tile `it` resident for all waves

    BFrag a; load_a(a, &lx[buf][w * 16 * 32], 32, lane);
#pragma unroll
    for (int m = 0; m < 3; m++) {
#pragma unroll
      for (int nt = 0; nt < 4; nt++) {
        BFrag bf; load_b_kc(bf, &lw[buf][m][nt * 16 * 32], 32, lane);
        acc[m][nt] = wmma_bf16(a, bf, acc[m][nt]);
      }
    }
    __syncthreads();                       // done reading buf before reuse
  }
  int col = lane & 15;
  u16t* outs[2] = { qb, kb };
#pragma unroll
  for (int m = 0; m < 2; m++) {
    float scl = (m == 0) ? 0.125f : 1.0f;   // fold 1/sqrt(HD) into Q
#pragma unroll
    for (int nt = 0; nt < 4; nt++)
#pragma unroll
      for (int r = 0; r < 8; r++) {
        int row = (lane < 16) ? r : (8 + r);
        int s = s0 + w * 16 + row;
        outs[m][(size_t)(bh * S_ + s) * HD_ + nt * 16 + col] =
            f32_bf16(acc[m][nt][r] * scl);
      }
  }
#pragma unroll
  for (int nt = 0; nt < 4; nt++)   // V stored transposed [bh][e][s]
#pragma unroll
    for (int r = 0; r < 8; r++) {
      int row = (lane < 16) ? r : (8 + r);
      vbT[((size_t)bh * HD_ + nt * 16 + col) * S_ + s0 + w * 16 + row] =
          f32_bf16(acc[2][nt][r]);
    }
}

// ---------------- Stage 2: flash attention ----------------
// grid (B*H, S/128), 256 threads; wave w -> query tile qt = by*8+w (16 rows).
__global__ __launch_bounds__(256) void k_flash(
    const u16t* __restrict__ qb, const u16t* __restrict__ kb,
    const u16t* __restrict__ vbT, u16t* __restrict__ ab) {
  __shared__ u16t lp[8][16 * 32];   // wave-private P staging
  int t = threadIdx.x, lane = t & 31, w = t >> 5;
  int bh = blockIdx.x, h = bh & 15, b = bh >> 4;
  int qt = blockIdx.y * 8 + w;
  int qbase = qt * 16;
  int col = lane & 15;

  const u16t* qptr = qb + (size_t)(bh * S_ + qbase) * HD_;
  BFrag qA[2];
  load_a(qA[0], qptr, HD_, lane);        // e = 0..31
  load_a(qA[1], qptr + 32, HD_, lane);   // e = 32..63

  v8f acc[4];
#pragma unroll
  for (int nt = 0; nt < 4; nt++) acc[nt] = vzero();
  float mrow[8], lrow[8];
#pragma unroll
  for (int r = 0; r < 8; r++) { mrow[r] = -1e30f; lrow[r] = 0.f; }

  int j2max = qt >> 1;   // causal: 32-wide key chunks up to the diagonal
  for (int j2 = 0; j2 <= j2max; j2++) {
    int kc0 = j2 * 32;
    const u16t* kptr = kb + (size_t)(bh * S_ + kc0) * HD_;
    v8f st[2];
#pragma unroll
    for (int tt = 0; tt < 2; tt++) {   // two 16-wide score tiles
      BFrag b0, b1;
      load_b_kc(b0, kptr + tt * 16 * HD_, HD_, lane);        // K^T, e=0..31
      load_b_kc(b1, kptr + tt * 16 * HD_ + 32, HD_, lane);   // e=32..63
      st[tt] = wmma_bf16(qA[0], b0, vzero());
      st[tt] = wmma_bf16(qA[1], b1, st[tt]);
    }
    float alpha[8];
#pragma unroll
    for (int r = 0; r < 8; r++) {
      int row = (lane < 16) ? r : (8 + r);
      int rg = qbase + row;
      float s0v = st[0][r]; if (kc0 + col > rg)      s0v = -1e30f;
      float s1v = st[1][r]; if (kc0 + 16 + col > rg) s1v = -1e30f;
      float mx = redmax16(fmaxf(s0v, s1v));
      float mn = fmaxf(mrow[r], mx);
      alpha[r] = __expf(mrow[r] - mn);
      float p0 = __expf(s0v - mn), p1 = __expf(s1v - mn);
      lrow[r] = lrow[r] * alpha[r] + redsum16(p0 + p1);
      mrow[r] = mn;
      lp[w][row * 32 + col]      = f32_bf16(p0);
      lp[w][row * 32 + 16 + col] = f32_bf16(p1);
    }
#pragma unroll
    for (int nt = 0; nt < 4; nt++)
#pragma unroll
      for (int r = 0; r < 8; r++) acc[nt][r] *= alpha[r];
    asm volatile("s_wait_dscnt 0" ::: "memory");   // P stores -> P loads
    BFrag pA; load_a(pA, &lp[w][0], 32, lane);     // P as 16x32 A-matrix
    const u16t* vptr = vbT + (size_t)bh * HD_ * S_ + kc0;   // [e][s], K-contiguous
#pragma unroll
    for (int nt = 0; nt < 4; nt++) {
      BFrag bv; load_b_kc(bv, vptr + (size_t)(nt * 16) * S_, S_, lane);
      acc[nt] = wmma_bf16(pA, bv, acc[nt]);
    }
  }
#pragma unroll
  for (int nt = 0; nt < 4; nt++)
#pragma unroll
    for (int r = 0; r < 8; r++) {
      int row = (lane < 16) ? r : (8 + r);
      float o = acc[nt][r] / lrow[r];
      ab[(size_t)(b * S_ + qbase + row) * (H_ * HD_) + h * HD_ + nt * 16 + col] =
          f32_bf16(o);
    }
}

// ---------------- Stage 3: output projection + bias ----------------
// grid (B*S/128, D/64), 256 threads. Wo pre-transposed [n][k].
// Same double-buffered async pipeline (3 async ops per thread per tile).
__global__ __launch_bounds__(256) void k_proj(
    const u16t* __restrict__ ab, const u16t* __restrict__ woT,
    const float* __restrict__ bo, float* __restrict__ out) {
  __shared__ u16t la[2][128 * 32];   // 16 KB
  __shared__ u16t lwT[2][64 * 32];   // 8 KB
  int t = threadIdx.x, lane = t & 31, w = t >> 5;
  int r0 = blockIdx.x * 128;
  int n0 = blockIdx.y * 64;
  v8f acc[4];
#pragma unroll
  for (int nt = 0; nt < 4; nt++) acc[nt] = vzero();

  int ar = t >> 1, ac = (t & 1) * 16;
  int f0 = t * 8, wn = f0 >> 5, wk = f0 & 31;

  auto issue = [&](int k0, int buf) {          // 3 async ops per thread
    const u16t* asrc = ab + (size_t)(r0 + ar) * D_ + k0 + ac;
    cp16(&la[buf][ar * 32 + ac], asrc);
    cp16(&la[buf][ar * 32 + ac + 8], asrc + 8);
    cp16(&lwT[buf][wn * 32 + wk], woT + (size_t)(n0 + wn) * D_ + k0 + wk);
  };

  constexpr int NIT = D_ / 32;
  issue(0, 0);
  for (int it = 0; it < NIT; it++) {
    int buf = it & 1;
    if (it + 1 < NIT) { issue((it + 1) * 32, (it + 1) & 1); ASYNC_WAIT(3); }
    else              { ASYNC_WAIT(0); }
    __syncthreads();

    BFrag a; load_a(a, &la[buf][w * 16 * 32], 32, lane);
#pragma unroll
    for (int nt = 0; nt < 4; nt++) {
      BFrag bf; load_b_kc(bf, &lwT[buf][nt * 16 * 32], 32, lane);
      acc[nt] = wmma_bf16(a, bf, acc[nt]);
    }
    __syncthreads();
  }
  int col = lane & 15;
#pragma unroll
  for (int nt = 0; nt < 4; nt++)
#pragma unroll
    for (int r = 0; r < 8; r++) {
      int row = (lane < 16) ? r : (8 + r);
      out[(size_t)(r0 + w * 16 + row) * D_ + n0 + nt * 16 + col] =
          acc[nt][r] + bo[n0 + nt * 16 + col];
    }
}

extern "C" void kernel_launch(void* const* d_in, const int* in_sizes, int n_in,
                              void* d_out, int out_size, void* d_ws, size_t ws_size,
                              hipStream_t stream) {
  const float* x  = (const float*)d_in[0];
  const float* Wq = (const float*)d_in[1];
  const float* Wk = (const float*)d_in[2];
  const float* Wv = (const float*)d_in[3];
  const float* Wo = (const float*)d_in[4];
  const float* bo = (const float*)d_in[5];
  (void)in_sizes; (void)n_in; (void)out_size; (void)ws_size;

  char* ws = (char*)d_ws;
  size_t off = 0;
  auto grab = [&](size_t nelem) {
    u16t* p = (u16t*)(ws + off); off += nelem * sizeof(u16t); return p;
  };
  u16t* xb   = grab((size_t)B_ * S_ * D_);        // 8 MB
  u16t* wqT  = grab((size_t)H_ * D_ * HD_);       // 2 MB, [h][e][d]
  u16t* wkT  = grab((size_t)H_ * D_ * HD_);
  u16t* wvT  = grab((size_t)H_ * D_ * HD_);
  u16t* woT  = grab((size_t)D_ * D_);             // 2 MB, [n][k]
  u16t* qb   = grab((size_t)B_ * H_ * S_ * HD_);  // 8 MB
  u16t* kb   = grab((size_t)B_ * H_ * S_ * HD_);
  u16t* vbT  = grab((size_t)B_ * H_ * S_ * HD_);  // 8 MB, [bh][e][s]
  u16t* abuf = grab((size_t)B_ * S_ * H_ * HD_);  // 8 MB  (total ~48 MB)

  int nx = B_ * S_ * D_, nw = H_ * D_ * HD_, nwo = D_ * D_;
  k_cvt  <<<nx / 256, 256, 0, stream>>>(x, xb, nx);
  k_cvt_t<<<nw / 256, 256, 0, stream>>>(Wq, wqT, D_, HD_, nw);
  k_cvt_t<<<nw / 256, 256, 0, stream>>>(Wk, wkT, D_, HD_, nw);
  k_cvt_t<<<nw / 256, 256, 0, stream>>>(Wv, wvT, D_, HD_, nw);
  k_cvt_t<<<nwo / 256, 256, 0, stream>>>(Wo, woT, D_, D_, nwo);

  k_qkv  <<<dim3(B_ * H_, S_ / 128), 256, 0, stream>>>(xb, wqT, wkT, wvT, qb, kb, vbT);
  k_flash<<<dim3(B_ * H_, S_ / 128), 256, 0, stream>>>(qb, kb, vbT, abuf);
  k_proj <<<dim3(B_ * S_ / 128, D_ / 64), 256, 0, stream>>>(abuf, woT, bo, (float*)d_out);
}
